// ConvTemporalGraphical_53102975648379
// MI455X (gfx1250) — matrix operand order repaired and checked by reading, not compiled, not run on hardware
//
#include <hip/hip_runtime.h>
#include <hip/hip_bf16.h>

// ---------------------------------------------------------------------------
// ST-GCN ConvTemporalGraphical, fused:  out[n,c,t,w] = sum_k (W_k x + b_k) A_k
// N=16, C_in=C_out=128, K=3, T=256, V=64.
// Per (n,t) slice: Y_k = W_k (128x128) * X (128x64);  out += Y_k * A_k (64x64)
// bf16 WMMA (v_wmma_f32_16x16x32_bf16), f32 accumulate. Intermediate Y never
// leaves the WGP (registers -> wave-private LDS strip). Bias folded into the
// epilogue via column sums of A_k.
//
// v4 (= v2): k-loop at unroll(1); compiler keeps the 16 XT B-fragments as a
//     128-VGPR register tile across the k-loop (read from LDS once per slice)
//     -> ~254 VGPRs, 4 waves/SIMD, 2 blocks/WGP. That is the intended
//     matrix-core design point. NOTE: do NOT add a waves-per-EU minimum here;
//     v3 showed the compiler then spills the hoisted fragments to scratch
//     ("Folded Reload" scratch_load_b128) instead of re-reading LDS.
// ---------------------------------------------------------------------------

typedef __attribute__((ext_vector_type(16))) __bf16 v16bf;
typedef __attribute__((ext_vector_type(8)))  float  v8f;

constexpr int CIN  = 128;
constexpr int COUT = 128;
constexpr int KK   = 3;
constexpr int TT   = 256;
constexpr int VV   = 64;
constexpr int NN   = 16;

constexpr int XT_S = 144;  // padded LDS stride for X^T (bf16 elems): breaks bank aliasing
constexpr int Y_S  = 72;   // padded LDS stride for Y strips
constexpr int TILES_PER_BLOCK = 4;

union FragBF { v16bf v; uint4 q[2]; };

// ---------------------------------------------------------------------------
// Prep: W (fp32) -> bf16 row-major; A (fp32) -> bf16 transposed AT[k][w][v];
// scol[k*64+w] = sum_v A[k][v][w]  (bias fold:  (b_k 1^T) A_k = b_k (x) scol_k)
// ---------------------------------------------------------------------------
__global__ __launch_bounds__(256)
void stgcn_prep(const float* __restrict__ A, const float* __restrict__ W,
                __bf16* __restrict__ wbf,
                __bf16* __restrict__ atbf,
                float* __restrict__ scol) {
  const int i = blockIdx.x * 256 + threadIdx.x;
  if (i < KK * COUT * CIN) wbf[i] = (__bf16)W[i];
  if (i < KK * VV * VV) {
    const int k = i / (VV * VV);
    const int rem = i - k * VV * VV;
    const int v = rem >> 6, w = rem & 63;
    atbf[(k * VV + w) * VV + v] = (__bf16)A[i];     // transpose
  }
  if (i < KK * VV) {
    const int k = i >> 6, w = i & 63;
    float s = 0.f;
    for (int v = 0; v < VV; ++v) s += A[(k * VV + v) * VV + w];
    scol[i] = s;
  }
}

// ---------------------------------------------------------------------------
// Main fused kernel: 256 threads = 8 wave32; each wave owns a 16-row c-strip.
// ---------------------------------------------------------------------------
__global__ __launch_bounds__(256)
void stgcn_fused(const float* __restrict__ x, const float* __restrict__ b,
                 const __bf16* __restrict__ wbf,
                 const __bf16* __restrict__ atbf,
                 const float* __restrict__ scol,
                 float* __restrict__ out) {
  __shared__ __align__(16) __bf16 XT[VV * XT_S];       // X^T, bf16: [v][cin]
  __shared__ __align__(16) __bf16 Yl[8 * 16 * Y_S];    // per-wave Y strips

  const int tid   = threadIdx.x;
  const int lane  = tid & 31;
  const int wave  = tid >> 5;
  const int lhalf = lane >> 4;     // 0: lanes 0-15, 1: lanes 16-31
  const int lrow  = lane & 15;
  const int c0    = wave * 16;
  __bf16* Yw = &Yl[wave * 16 * Y_S];

  for (int it = 0; it < TILES_PER_BLOCK; ++it) {
    const int tile = blockIdx.x * TILES_PER_BLOCK + it;
    const int n  = tile >> 8;          // T == 256
    const int tt = tile & 255;
    const float* xbase = x + (size_t)n * CIN * TT * VV + (size_t)tt * VV;

    // Speculative prefetch of next slice (global_prefetch_b8; safe if dropped).
    if (it + 1 < TILES_PER_BLOCK) {
      const float* xn = xbase + VV;
      __builtin_prefetch(xn + (size_t)(tid >> 1) * (TT * VV) + (tid & 1) * 32, 0, 0);
    }

    // ---- Stage X slice (128 x 64 fp32) into LDS transposed as bf16 --------
    {
      const int vg = (tid & 15) * 4;   // 4 consecutive v per thread
      const int cb = tid >> 4;
      #pragma unroll
      for (int p = 0; p < 8; ++p) {
        const int cin = cb + p * 16;
        const float4 f = *(const float4*)(xbase + (size_t)cin * (TT * VV) + vg);
        XT[(vg + 0) * XT_S + cin] = (__bf16)f.x;
        XT[(vg + 1) * XT_S + cin] = (__bf16)f.y;
        XT[(vg + 2) * XT_S + cin] = (__bf16)f.z;
        XT[(vg + 3) * XT_S + cin] = (__bf16)f.w;
      }
    }
    __syncthreads();

    v8f acc[4] = {};   // final 16x64 strip (4 col tiles), f32

    // unroll(1): keep only one k-iteration's W fragments / yacc live; the XT
    // fragments become a register tile reused across all 3 k-iterations.
    #pragma unroll 1
    for (int k = 0; k < KK; ++k) {
      // ---- Stage 1: Y_k strip = W_k[c0..c0+15, :] * X  ---------------------
      v8f yacc[4] = {};
      const __bf16* wrow = wbf + (size_t)(k * COUT + c0 + lrow) * CIN;
      const int klo = lhalf * 8;       // A-frag: {klo..klo+7},{klo+16..klo+23}
      #pragma unroll
      for (int kc = 0; kc < 4; ++kc) {          // K = 128 in chunks of 32
        const int kb = kc * 32;
        FragBF af;
        af.q[0] = *(const uint4*)(wrow + kb + klo);
        af.q[1] = *(const uint4*)(wrow + kb + klo + 16);
        #pragma unroll
        for (int t = 0; t < 4; ++t) {           // 4 v-tiles of 16
          FragBF bfr;                           // B-frag: {klo..klo+7},{klo+16..klo+23}
          const __bf16* xr = &XT[(t * 16 + lrow) * XT_S + kb + lhalf * 8];
          bfr.q[0] = *(const uint4*)(xr);
          bfr.q[1] = *(const uint4*)(xr + 16);
          yacc[t] = __builtin_amdgcn_wmma_f32_16x16x32_bf16(
              false, af.v, false, bfr.v, (short)0, yacc[t], false, false);
        }
      }

      // ---- Spill Y strip (16x64) to wave-private LDS as bf16 --------------
      // C/D layout: VGPR r -> row r + 8*lhalf, col = t*16 + lrow. Same-wave DS
      // ops are in-order, so no barrier needed before re-reading below.
      #pragma unroll
      for (int t = 0; t < 4; ++t)
        #pragma unroll
        for (int r = 0; r < 8; ++r)
          Yw[(r + lhalf * 8) * Y_S + t * 16 + lrow] = (__bf16)yacc[t][r];

      // ---- Stage 2: acc += Y_k strip (16x64) * A_k (64x64) ----------------
      #pragma unroll
      for (int k2 = 0; k2 < 2; ++k2) {          // K = 64 in chunks of 32
        const int vb = k2 * 32;
        FragBF af2;
        const __bf16* yr = &Yw[lrow * Y_S + vb + lhalf * 8];
        af2.q[0] = *(const uint4*)(yr);
        af2.q[1] = *(const uint4*)(yr + 16);
        #pragma unroll
        for (int t = 0; t < 4; ++t) {           // 4 w-tiles of 16
          FragBF bf2;                           // AT[k][w][v]: contiguous in v
          const __bf16* ar =
              atbf + (size_t)(k * VV + t * 16 + lrow) * VV + vb + lhalf * 8;
          bf2.q[0] = *(const uint4*)(ar);
          bf2.q[1] = *(const uint4*)(ar + 16);
          acc[t] = __builtin_amdgcn_wmma_f32_16x16x32_bf16(
              false, af2.v, false, bf2.v, (short)0, acc[t], false, false);
        }
      }
    }

    // ---- Epilogue: fold bias, store out[n, c, tt, w] -----------------------
    float* obase = out + (size_t)n * COUT * TT * VV + (size_t)tt * VV;
    #pragma unroll
    for (int t = 0; t < 4; ++t) {
      const int w = t * 16 + lrow;
      float s0 = scol[0 * 64 + w], s1 = scol[1 * 64 + w], s2 = scol[2 * 64 + w];
      #pragma unroll
      for (int r = 0; r < 8; ++r) {
        const int c = c0 + lhalf * 8 + r;
        float v = acc[t][r];
        v += b[0 * 128 + c] * s0 + b[1 * 128 + c] * s1 + b[2 * 128 + c] * s2;
        obase[(size_t)c * (TT * VV) + w] = v;
      }
    }
    __syncthreads();   // XT reused next iteration
  }
}

// ---------------------------------------------------------------------------
extern "C" void kernel_launch(void* const* d_in, const int* in_sizes, int n_in,
                              void* d_out, int out_size, void* d_ws, size_t ws_size,
                              hipStream_t stream) {
  const float* x = (const float*)d_in[0];   // (16,128,256,64)
  const float* A = (const float*)d_in[1];   // (3,64,64)
  const float* W = (const float*)d_in[2];   // (384,128)
  const float* b = (const float*)d_in[3];   // (384,)
  float* out = (float*)d_out;               // (16,128,256,64)

  // Workspace carve-up (~124 KB): W bf16 | AT bf16 | colsums
  __bf16* wbf  = (__bf16*)d_ws;                        // 98304 B
  __bf16* atbf = wbf + KK * COUT * CIN;                // 24576 B
  float* scol = (float*)(atbf + KK * VV * VV);         //   768 B

  stgcn_prep<<<(KK * COUT * CIN + 255) / 256, 256, 0, stream>>>(A, W, wbf, atbf, scol);

  const int nblocks = (NN * TT) / TILES_PER_BLOCK;     // 4096 (n,t) slices / 4
  stgcn_fused<<<nblocks, 256, 0, stream>>>(x, b, wbf, atbf, scol, out);
}
